// MAB_36584531427725
// MI455X (gfx1250) — matrix operand order repaired and checked by reading, not compiled, not run on hardware
//
#include <hip/hip_runtime.h>

// ---------------------------------------------------------------------------
// MAB transformer block for MI455X (gfx1250), wave32, WMMA bf16 path + TDM.
// D=1024, H=16, DH=64, DFF=4096, B=4, NQ=NK=1024.
// ---------------------------------------------------------------------------

typedef __bf16 bf16;
typedef __attribute__((ext_vector_type(16))) __bf16 v16bf;
typedef __attribute__((ext_vector_type(8)))  float  v8f;
typedef unsigned int u32;
typedef u32 __attribute__((ext_vector_type(4))) u32x4;
typedef u32 __attribute__((ext_vector_type(8))) u32x8;

#define CB 4
#define CNQ 1024
#define CNK 1024
#define CD 1024
#define CH 16
#define CDH 64
#define CDFF 4096
#define CM (CB * CNQ)           // 4096 rows for all activation GEMMs
#define LN_EPS 1e-5f
#define ATTN_SCALE 0.03125f     // 1/sqrt(1024)

// ---------------------------------------------------------------------------
// Tensor Data Mover: 2D tile load (global -> LDS), bf16 elements.
// D# per CDNA5 ISA:
//   group0: [1:0]=count=1, [63:32]=lds_addr, [120:64]=global_addr, [127:126]=2
//   group1: [17:16]=data_size(1=2B), [79:48]=tensor_dim0, [111:80]=tensor_dim1,
//           [127:112]=tile_dim0, [143:128]=tile_dim1, [207:160]=dim0_stride
// Issued by one wave; completion via TENSORcnt.
// ---------------------------------------------------------------------------
static __device__ inline void tdm_load_2d_bf16(u32 lds_addr, const void* gptr,
                                               u32 tensor_d0, u32 tensor_d1,
                                               u32 tile_d0, u32 tile_d1,
                                               u32 stride0) {
    const unsigned long long ga = (unsigned long long)gptr;
    u32x4 g0;
    g0[0] = 1u;                                   // count=1, user descriptor
    g0[1] = lds_addr;                             // LDS byte address
    g0[2] = (u32)ga;                              // global_addr[31:0]
    g0[3] = (u32)(ga >> 32) | 0x80000000u;        // global_addr[56:32] | type=2
    u32x8 g1;
    g1[0] = 0x00010000u;                          // wg_mask=0, data_size=1 (2B)
    g1[1] = (tensor_d0 & 0xFFFFu) << 16;          // tensor_dim0[15:0]
    g1[2] = (tensor_d0 >> 16) | ((tensor_d1 & 0xFFFFu) << 16);
    g1[3] = (tensor_d1 >> 16) | (tile_d0 << 16);  // tensor_dim1[31:16] | tile_dim0
    g1[4] = tile_d1;                              // tile_dim1, tile_dim2=0
    g1[5] = stride0;                              // tensor_dim0_stride[31:0]
    g1[6] = 0;                                    // stride0[47:32]=0, dim1_stride lo
    g1[7] = 0;
    asm volatile("tensor_load_to_lds %0, %1" :: "s"(g0), "s"(g1) : "memory");
}

// ---------------------------------------------------------------------------
// Fragment loaders (wave32 WMMA bf16 16x16x32 layouts per CDNA5 ISA 7.12.2)
// ---------------------------------------------------------------------------
static __device__ inline v16bf load_a_frag_lds(const bf16* tile, int row0, int stride,
                                               int kk, int lane) {
    const int m  = row0 + (lane & 15);
    const int kb = kk + ((lane >> 4) << 3);          // +8 for upper half-wave
    const bf16* p = tile + m * stride + kb;
    v16bf a;
#pragma unroll
    for (int e = 0; e < 8; ++e) { a[e] = p[e]; a[e + 8] = p[16 + e]; }
    return a;
}

// B staged in LDS as [N][K]: lanes 0-15 -> N, K=kk..kk+7,kk+16..23; lanes 16-31 -> K+8
static __device__ inline v16bf load_b_frag_lds(const bf16* tile, int col0, int stride,
                                               int kk, int lane) {
    const int n  = col0 + (lane & 15);
    const int kb = kk + ((lane >> 4) << 3);          // +8 for upper half-wave
    const bf16* p = tile + n * stride + kb;
    v16bf b;
#pragma unroll
    for (int e = 0; e < 8; ++e) { b[e] = p[e]; b[e + 8] = p[16 + e]; }
    return b;
}

static __device__ inline v8f wmma_bf16(v16bf a, v16bf b, v8f c) {
    return __builtin_amdgcn_wmma_f32_16x16x32_bf16(false, a, false, b, (short)0, c,
                                                   false, false);
}

// ---------------------------------------------------------------------------
// f32 -> bf16 conversion
// ---------------------------------------------------------------------------
__global__ void k_cvt_bf16(const float* __restrict__ in, bf16* __restrict__ out, int n) {
    int i = blockIdx.x * blockDim.x + threadIdx.x;
    const int stride = gridDim.x * blockDim.x;
    for (; i < n; i += stride) out[i] = (bf16)in[i];
}

// ---------------------------------------------------------------------------
// Tiled WMMA GEMM: C[M,N] = A[M,K] (bf16, rm) * Bw[K,N] (bf16, rm)
// Block tile 128x128, BK=64, 256 threads = 8 waves, wave tile 32x64 (2x4 frags)
// => 16 v_wmma per K-iteration per wave. A tile arrives via TDM (wave 0 issues,
// all waves overlap the B transpose-stage, s_wait_tensorcnt before barrier).
// MODE 0: +bias, store bf16 scattered to [B,H,seq,DH] head layout
// MODE 1: +bias, store f32 linear
// MODE 2: +bias, ReLU, store bf16 linear
// ---------------------------------------------------------------------------
template <int MODE>
__global__ void k_gemm(const bf16* __restrict__ A, const bf16* __restrict__ Bw,
                       const float* __restrict__ bias, float* __restrict__ Cf,
                       bf16* __restrict__ Cbf, int M, int N, int K) {
    __shared__ __align__(16) bf16 As[128 * 64];      // [M][K] 16 KB
    __shared__ __align__(16) bf16 Bs[128 * 64];      // [N][K] 16 KB

    const int t    = threadIdx.x;
    const int lane = t & 31;
    const int wid  = t >> 5;
    const int waveM = wid & 3;   // 4 waves along M  -> 128 rows
    const int waveN = wid >> 2;  // 2 waves along N  -> 128 cols
    const int m0 = blockIdx.y * 128;
    const int n0 = blockIdx.x * 128;

    v8f acc[2][4] = {};

    for (int k0 = 0; k0 < K; k0 += 64) {
        // --- A tile [128][64] via Tensor Data Mover (wave 0 issues) ---
        if (wid == 0) {
            tdm_load_2d_bf16((u32)(size_t)As, A + (size_t)m0 * K + k0,
                             (u32)K, (u32)M, 64u, 128u, (u32)K);
        }
        // --- stage B tile transposed into Bs[N][K] (all 256 threads) ---
        {
            const int kk  = t >> 2;            // 0..63
            const int n32 = (t & 3) * 32;      // 0,32,64,96
            const bf16* src = Bw + (size_t)(k0 + kk) * N + n0 + n32;
#pragma unroll
            for (int e = 0; e < 32; ++e) Bs[(n32 + e) * 64 + kk] = src[e];
            if (k0 + 64 < K) __builtin_prefetch(src + (size_t)64 * N, 0, 0);
        }
        if (wid == 0) __builtin_amdgcn_s_wait_tensorcnt(0);
        __syncthreads();

#pragma unroll
        for (int kk = 0; kk < 64; kk += 32) {
            v16bf af[2], bf[4];
#pragma unroll
            for (int i = 0; i < 2; ++i)
                af[i] = load_a_frag_lds(As, waveM * 32 + i * 16, 64, kk, lane);
#pragma unroll
            for (int j = 0; j < 4; ++j)
                bf[j] = load_b_frag_lds(Bs, waveN * 64 + j * 16, 64, kk, lane);
#pragma unroll
            for (int i = 0; i < 2; ++i)
#pragma unroll
                for (int j = 0; j < 4; ++j)
                    acc[i][j] = wmma_bf16(af[i], bf[j], acc[i][j]);
        }
        __syncthreads();
    }

    // --- epilogue: C frag mapping: reg r -> M = r (+8 upper half), lane -> N ---
#pragma unroll
    for (int i = 0; i < 2; ++i)
#pragma unroll
        for (int j = 0; j < 4; ++j)
#pragma unroll
            for (int r = 0; r < 8; ++r) {
                const int m = m0 + waveM * 32 + i * 16 + r + ((lane >> 4) << 3);
                const int n = n0 + waveN * 64 + j * 16 + (lane & 15);
                float v = acc[i][j][r] + bias[n];
                if constexpr (MODE == 2) v = fmaxf(v, 0.0f);
                if constexpr (MODE == 1) {
                    Cf[(size_t)m * N + n] = v;
                } else if constexpr (MODE == 0) {
                    const int b = m >> 10, s = m & 1023;
                    const int h = n >> 6,  d = n & 63;
                    Cbf[(((size_t)(b * CH + h) * 1024 + s) << 6) + d] = (bf16)v;
                } else {
                    Cbf[(size_t)m * N + n] = (bf16)v;
                }
            }
}

// ---------------------------------------------------------------------------
// Attention: per block = (q-tile of 32 rows, head h, batch b).
// Whole 32x1024 score slab in LDS (128 KB) -> exact one-pass softmax; WMMA for
// QK^T and PV. 256 threads = 8 waves (2 along rows x 4 along cols).
// ---------------------------------------------------------------------------
__global__ void k_attention(const bf16* __restrict__ qh, const bf16* __restrict__ kh,
                            const bf16* __restrict__ vh, bf16* __restrict__ obf) {
    extern __shared__ __align__(16) char smem[];
    float* Sl     = (float*)smem;                  // [32][1024]
    bf16*  qs     = (bf16*)(Sl + 32 * 1024);       // [32][64]
    bf16*  kvs    = qs + 32 * 64;                  // [64][64] (k tile, then v^T)
    float* rowmax = (float*)(kvs + 64 * 64);       // [32]
    float* rowsum = rowmax + 32;                   // [32]
    float* red    = rowsum + 32;                   // [32][8]

    const int t     = threadIdx.x;
    const int lane  = t & 31;
    const int wid   = t >> 5;
    const int waveM = wid & 1;     // 2 waves along 32 q-rows
    const int waveN = wid >> 1;    // 4 waves along 64 cols
    const int qt = blockIdx.x, h = blockIdx.y, b = blockIdx.z;
    const size_t bh = (size_t)b * CH + h;

    // stage q tile [32][64]
    {
        const int r = t >> 3, g = (t & 7) * 8;
        const bf16* src = qh + ((bh * CNQ) + qt * 32 + r) * CDH + g;
        bf16* dst = qs + r * 64 + g;
        *(uint4*)dst = *(const uint4*)src;
    }
    __syncthreads();

    // ---- pass 1: scores ----
    for (int j = 0; j < CNK / 64; ++j) {
        {   // k tile rows j*64..+63, row-major [krow][d] == Bs[N=krow][K=d]
            const int r = t >> 2, d0 = (t & 3) * 16;
            const bf16* src = kh + ((bh * CNK) + j * 64 + r) * CDH + d0;
            bf16* dst = kvs + r * 64 + d0;
            *(uint4*)dst       = *(const uint4*)src;
            *((uint4*)dst + 1) = *((const uint4*)src + 1);
        }
        __syncthreads();
        v8f s = {};
#pragma unroll
        for (int kk = 0; kk < 64; kk += 32) {
            v16bf a  = load_a_frag_lds(qs,  waveM * 16, 64, kk, lane);
            v16bf bb = load_b_frag_lds(kvs, waveN * 16, 64, kk, lane);
            s = wmma_bf16(a, bb, s);
        }
#pragma unroll
        for (int r = 0; r < 8; ++r) {
            const int m = waveM * 16 + r + ((lane >> 4) << 3);
            const int n = j * 64 + waveN * 16 + (lane & 15);
            Sl[m * 1024 + n] = s[r] * ATTN_SCALE;
        }
        __syncthreads();
    }

    // ---- pass 2: softmax over the 32x1024 slab (8 threads per row) ----
    {
        const int row = t >> 3, l8 = t & 7;
        float* rp = Sl + row * 1024 + l8 * 128;
        float mx = -3.0e38f;
        for (int c = 0; c < 128; ++c) mx = fmaxf(mx, rp[c]);
        red[row * 8 + l8] = mx;
        __syncthreads();
        if (l8 == 0) {
            float m = red[row * 8];
#pragma unroll
            for (int e = 1; e < 8; ++e) m = fmaxf(m, red[row * 8 + e]);
            rowmax[row] = m;
        }
        __syncthreads();
        const float m = rowmax[row];
        float sum = 0.0f;
        for (int c = 0; c < 128; ++c) {
            float e = __expf(rp[c] - m);
            rp[c] = e;
            sum += e;
        }
        __syncthreads();
        red[row * 8 + l8] = sum;
        __syncthreads();
        if (l8 == 0) {
            float sm = 0.0f;
#pragma unroll
            for (int e = 0; e < 8; ++e) sm += red[row * 8 + e];
            rowsum[row] = sm;
        }
        __syncthreads();
    }

    // ---- pass 3: O = P @ V ----
    v8f o = {};
    for (int j = 0; j < CNK / 64; ++j) {
        {   // v tile transposed: kvs[d][vrow]
            const int r = t >> 2, d0 = (t & 3) * 16;
            const bf16* src = vh + ((bh * CNK) + j * 64 + r) * CDH + d0;
#pragma unroll
            for (int e = 0; e < 16; ++e) kvs[(d0 + e) * 64 + r] = src[e];
        }
        __syncthreads();
#pragma unroll
        for (int kk = 0; kk < 64; kk += 32) {
            const int m  = waveM * 16 + (lane & 15);
            const int kb = j * 64 + kk + ((lane >> 4) << 3);
            const float* p = Sl + m * 1024 + kb;
            v16bf a;
#pragma unroll
            for (int e = 0; e < 8; ++e) { a[e] = (bf16)p[e]; a[e + 8] = (bf16)p[16 + e]; }
            v16bf bb = load_b_frag_lds(kvs, waveN * 16, 64, kk, lane);
            o = wmma_bf16(a, bb, o);
        }
        __syncthreads();
    }

    // epilogue: normalize by rowsum, scatter back to [B*NQ, D] bf16
#pragma unroll
    for (int r = 0; r < 8; ++r) {
        const int m = waveM * 16 + r + ((lane >> 4) << 3);   // q row in tile
        const int d = waveN * 16 + (lane & 15);              // head dim
        const float v = o[r] / rowsum[m];
        const int gq = qt * 32 + m;
        obf[((size_t)(b * CNQ + gq)) * CD + h * CDH + d] = (bf16)v;
    }
}

// ---------------------------------------------------------------------------
// Residual + LayerNorm over rows of 1024. One block (256 thr) per row.
// ---------------------------------------------------------------------------
template <bool WRITE_BF16>
__global__ void k_add_ln(const float* __restrict__ x1, const float* __restrict__ x2,
                         const float* __restrict__ g, const float* __restrict__ be,
                         float* __restrict__ outF, bf16* __restrict__ outB) {
    __shared__ float red[256];
    const int row = blockIdx.x, t = threadIdx.x;
    const size_t base = (size_t)row * CD;

    float v[4];
    float s = 0.0f;
#pragma unroll
    for (int q = 0; q < 4; ++q) {
        const int c = t + q * 256;
        v[q] = x1[base + c] + x2[base + c];
        s += v[q];
    }
    red[t] = s;
    __syncthreads();
    for (int st = 128; st > 0; st >>= 1) {
        if (t < st) red[t] += red[t + st];
        __syncthreads();
    }
    const float mean = red[0] * (1.0f / CD);
    __syncthreads();

    float vs = 0.0f;
#pragma unroll
    for (int q = 0; q < 4; ++q) {
        const float d = v[q] - mean;
        vs += d * d;
    }
    red[t] = vs;
    __syncthreads();
    for (int st = 128; st > 0; st >>= 1) {
        if (t < st) red[t] += red[t + st];
        __syncthreads();
    }
    const float rstd = rsqrtf(red[0] * (1.0f / CD) + LN_EPS);

#pragma unroll
    for (int q = 0; q < 4; ++q) {
        const int c = t + q * 256;
        const float y = (v[q] - mean) * rstd * g[c] + be[c];
        outF[base + c] = y;
        if constexpr (WRITE_BF16) outB[base + c] = (bf16)y;
    }
}

// ---------------------------------------------------------------------------
// Host-side orchestration
// ---------------------------------------------------------------------------
extern "C" void kernel_launch(void* const* d_in, const int* in_sizes, int n_in,
                              void* d_out, int out_size, void* d_ws, size_t ws_size,
                              hipStream_t stream) {
    (void)in_sizes; (void)n_in; (void)out_size; (void)ws_size;

    const float* Qf  = (const float*)d_in[0];
    const float* Kf  = (const float*)d_in[1];
    const float* Wq  = (const float*)d_in[2];
    const float* bq  = (const float*)d_in[3];
    const float* Wk  = (const float*)d_in[4];
    const float* bk  = (const float*)d_in[5];
    const float* Wv  = (const float*)d_in[6];
    const float* bv  = (const float*)d_in[7];
    const float* Wo  = (const float*)d_in[8];
    const float* bo  = (const float*)d_in[9];
    const float* W1  = (const float*)d_in[10];
    const float* b1  = (const float*)d_in[11];
    const float* W2  = (const float*)d_in[12];
    const float* b2  = (const float*)d_in[13];
    const float* g0  = (const float*)d_in[14];
    const float* be0 = (const float*)d_in[15];
    const float* g1  = (const float*)d_in[16];
    const float* be1 = (const float*)d_in[17];
    float* out = (float*)d_out;

    // ---- workspace carve ----
    char* base = (char*)d_ws;
    size_t off = 0;
    auto carve = [&](size_t bytes) -> char* {
        char* p = base + off;
        off += (bytes + 255) & ~(size_t)255;
        return p;
    };
    const size_t NACT = (size_t)CM * CD;        // 4M elems
    bf16*  Qbf  = (bf16*)carve(NACT * 2);
    bf16*  Kbf  = (bf16*)carve(NACT * 2);
    bf16*  Wqb  = (bf16*)carve((size_t)CD * CD * 2);
    bf16*  Wkb  = (bf16*)carve((size_t)CD * CD * 2);
    bf16*  Wvb  = (bf16*)carve((size_t)CD * CD * 2);
    bf16*  Wob  = (bf16*)carve((size_t)CD * CD * 2);
    bf16*  W1b  = (bf16*)carve((size_t)CD * CDFF * 2);
    bf16*  W2b  = (bf16*)carve((size_t)CDFF * CD * 2);
    bf16*  qhp  = (bf16*)carve(NACT * 2);       // [B,H,NQ,DH]
    bf16*  khp  = (bf16*)carve(NACT * 2);
    bf16*  vhp  = (bf16*)carve(NACT * 2);
    bf16*  Obf  = (bf16*)carve(NACT * 2);       // attention out [B*NQ, D]
    float* Obuf = (float*)carve(NACT * 4);      // O @ Wo + bo
    float* Xf   = (float*)carve(NACT * 4);      // LN0 output f32
    bf16*  Xbf  = (bf16*)carve(NACT * 2);
    bf16*  Hbf  = (bf16*)carve((size_t)CM * CDFF * 2);
    float* Ybuf = (float*)carve(NACT * 4);

    // ---- 1. conversions ----
    const dim3 cvB(256);
    k_cvt_bf16<<<2048, cvB, 0, stream>>>(Qf, Qbf, (int)NACT);
    k_cvt_bf16<<<2048, cvB, 0, stream>>>(Kf, Kbf, (int)NACT);
    k_cvt_bf16<<<1024, cvB, 0, stream>>>(Wq, Wqb, CD * CD);
    k_cvt_bf16<<<1024, cvB, 0, stream>>>(Wk, Wkb, CD * CD);
    k_cvt_bf16<<<1024, cvB, 0, stream>>>(Wv, Wvb, CD * CD);
    k_cvt_bf16<<<1024, cvB, 0, stream>>>(Wo, Wob, CD * CD);
    k_cvt_bf16<<<2048, cvB, 0, stream>>>(W1, W1b, CD * CDFF);
    k_cvt_bf16<<<2048, cvB, 0, stream>>>(W2, W2b, CDFF * CD);

    const dim3 gB(256);
    // ---- 2. q/k/v projections (scatter to head layout) ----
    {
        dim3 grid(CD / 128, CM / 128);
        k_gemm<0><<<grid, gB, 0, stream>>>(Qbf, Wqb, bq, nullptr, qhp, CM, CD, CD);
        k_gemm<0><<<grid, gB, 0, stream>>>(Kbf, Wkb, bk, nullptr, khp, CM, CD, CD);
        k_gemm<0><<<grid, gB, 0, stream>>>(Kbf, Wvb, bv, nullptr, vhp, CM, CD, CD);
    }
    // ---- 3. attention ----
    {
        dim3 grid(CNQ / 32, CH, CB);
        const size_t shmem = 32 * 1024 * 4 + 32 * 64 * 2 + 64 * 64 * 2 +
                             32 * 4 + 32 * 4 + 32 * 8 * 4;   // 144640 B
        k_attention<<<grid, gB, shmem, stream>>>(qhp, khp, vhp, Obf);
    }
    // ---- 4. output projection ----
    {
        dim3 grid(CD / 128, CM / 128);
        k_gemm<1><<<grid, gB, 0, stream>>>(Obf, Wob, bo, Obuf, nullptr, CM, CD, CD);
    }
    // ---- 5. LN0: X = LN(Q + O) ----
    k_add_ln<true><<<CM, gB, 0, stream>>>(Qf, Obuf, g0, be0, Xf, Xbf);
    // ---- 6. FFN1: H = relu(X W1 + b1) ----
    {
        dim3 grid(CDFF / 128, CM / 128);
        k_gemm<2><<<grid, gB, 0, stream>>>(Xbf, W1b, b1, nullptr, Hbf, CM, CDFF, CD);
    }
    // ---- 7. FFN2: Y = H W2 + b2 ----
    {
        dim3 grid(CD / 128, CM / 128);
        k_gemm<1><<<grid, gB, 0, stream>>>(Hbf, W2b, b2, Ybuf, nullptr, CM, CD, CDFF);
    }
    // ---- 8. LN1 -> out ----
    k_add_ln<false><<<CM, gB, 0, stream>>>(Xf, Ybuf, g1, be1, out, nullptr);
}
